// SingleSelfAttention_76742475645195
// MI455X (gfx1250) — compile-verified, matrix operand order checked by reading, not run on hardware
//
#include <hip/hip_runtime.h>
#include <hip/hip_bf16.h>

// ---------------------------------------------------------------------------
// Types for CDNA5 WMMA (wave32): A/B = 16 bf16 per lane, C/D = 8 f32 per lane
// ---------------------------------------------------------------------------
typedef __attribute__((ext_vector_type(16))) __bf16 v16bf;
typedef __attribute__((ext_vector_type(8)))  float  v8f;
typedef __attribute__((ext_vector_type(4)))  int    v4i;

#define LDS_STRIDE 40   // 32 + 8 pad (bf16 elems) to spread LDS banks

// Problem sizes
#define BB 16
#define SS 2048
#define DD 512
#define HH 512
#define MROWS (BB * SS)          // 32768
#define LN_EPS 1e-3f

// CDNA5 async global->LDS path (ASYNCcnt), guarded so compilation can never
// hard-fail: fall back to synchronous VGPR-mediated staging if the builtin
// is unavailable on this toolchain.
#if defined(__gfx1250__) && __has_builtin(__builtin_amdgcn_global_load_async_to_lds_b128) && \
    __has_builtin(__builtin_amdgcn_s_wait_asynccnt)
#define HAVE_ASYNC_LDS 1
#define GLOBAL_AS __attribute__((address_space(1)))
#define LDS_AS __attribute__((address_space(3)))
#else
#define HAVE_ASYNC_LDS 0
#endif

// ---------------------------------------------------------------------------
// Helpers
// ---------------------------------------------------------------------------
__device__ __forceinline__ float gelu_tanh(float x) {
  const float c = 0.7978845608028654f;   // sqrt(2/pi)
  float x3 = x * x * x;
  return 0.5f * x * (1.0f + tanhf(c * (x + 0.044715f * x3)));
}

// Load one 16x32 bf16 fragment (A or B^T) from LDS per CDNA5 WMMA layout:
// lane = half*16 + idx16 ; VGPR0..3 = K = half*8 + 0..7 ; VGPR4..7 = K+16.
__device__ __forceinline__ v16bf load_frag_lds(const __hip_bfloat16* base,
                                               int row, int half) {
  const char* p = (const char*)(base + row * LDS_STRIDE + half * 8);
  union { uint4 q[2]; v16bf v; } u;
  u.q[0] = *(const uint4*)p;          // K = half*8 .. half*8+7
  u.q[1] = *(const uint4*)(p + 32);   // K = 16+half*8 .. 16+half*8+7
  return u.v;
}

// Stage one 16B chunk global -> LDS (async if available)
__device__ __forceinline__ void stage16(const __hip_bfloat16* gsrc,
                                        __hip_bfloat16* ldst) {
#if HAVE_ASYNC_LDS
  __builtin_amdgcn_global_load_async_to_lds_b128(
      (GLOBAL_AS v4i*)(GLOBAL_AS void*)gsrc,
      (LDS_AS v4i*)(LDS_AS void*)ldst, 0, 0);
#else
  *(uint4*)ldst = *(const uint4*)gsrc;
#endif
}

// ---------------------------------------------------------------------------
// Kernel 1: f32 -> bf16 conversion (x)
// ---------------------------------------------------------------------------
__global__ void cvt_f32_bf16_kernel(const float* __restrict__ x,
                                    __hip_bfloat16* __restrict__ xb) {
  int i = blockIdx.x * blockDim.x + threadIdx.x;   // one float4 per thread
  float4 v = ((const float4*)x)[i];
  __hip_bfloat16 t[4];
  t[0] = __float2bfloat16(v.x);
  t[1] = __float2bfloat16(v.y);
  t[2] = __float2bfloat16(v.z);
  t[3] = __float2bfloat16(v.w);
  *(uint2*)(xb + (size_t)i * 4) = *(const uint2*)t;
}

// ---------------------------------------------------------------------------
// Kernel 2: transpose + convert weight  W[k][n] f32  ->  Wt[n][k] bf16
// ---------------------------------------------------------------------------
__global__ void transpose_cvt_kernel(const float* __restrict__ W,
                                     __hip_bfloat16* __restrict__ Wt) {
  __shared__ float tile[32][33];
  int nt = blockIdx.x, kt = blockIdx.y;
  int tx = threadIdx.x & 31, ty = threadIdx.x >> 5;   // ty 0..7
  #pragma unroll
  for (int i = 0; i < 4; ++i)
    tile[ty + i * 8][tx] = W[(size_t)(kt * 32 + ty + i * 8) * HH + nt * 32 + tx];
  __syncthreads();
  #pragma unroll
  for (int i = 0; i < 4; ++i)
    Wt[(size_t)(nt * 32 + ty + i * 8) * DD + kt * 32 + tx] =
        __float2bfloat16(tile[tx][ty + i * 8]);
}

// ---------------------------------------------------------------------------
// Kernel 3: fused  out = LayerNorm(gelu(A @ W + b))  via bf16 WMMA
//   A  : [32768, 512] bf16 row-major
//   Wt : [512, 512]  bf16, N-major (i.e. W transposed)
//   block = 256 threads (8 waves), tile = 64 M x 512 N (full LN row in block)
//   wave w owns N columns [w*64, w*64+64), 4x4 grid of 16x16 WMMA tiles.
//   Double-buffered LDS; async global->LDS staging overlapped with WMMA.
// ---------------------------------------------------------------------------
__global__ __launch_bounds__(256) void gemm_gelu_ln_kernel(
    const __hip_bfloat16* __restrict__ A,
    const __hip_bfloat16* __restrict__ Wt,
    const float* __restrict__ bias,
    __hip_bfloat16* __restrict__ out) {
  __shared__ __align__(16) __hip_bfloat16 As[2][64 * LDS_STRIDE];
  __shared__ __align__(16) __hip_bfloat16 Ws[2][512 * LDS_STRIDE];
  __shared__ float rsum[64];
  __shared__ float rssq[64];

  const int tid   = threadIdx.x;
  const int wave  = tid >> 5;
  const int lane  = tid & 31;
  const int half  = lane >> 4;
  const int col16 = lane & 15;
  const int m0    = blockIdx.x * 64;

  if (tid < 64) { rsum[tid] = 0.f; rssq[tid] = 0.f; }

  v8f acc[4][4];
  #pragma unroll
  for (int i = 0; i < 4; ++i)
    #pragma unroll
    for (int j = 0; j < 4; ++j)
      #pragma unroll
      for (int e = 0; e < 8; ++e) acc[i][j][e] = 0.f;

  const int arow = tid >> 2;   // 0..63
  const int akc  = tid & 3;    // 0..3

  // Stage tiles for K-step k0 into LDS buffer `b`.
  auto stage = [&](int b, int k0) {
    // A tile: 64 rows x 32 K (one b128 per thread)
    stage16(A + (size_t)(m0 + arow) * DD + k0 + akc * 8,
            &As[b][arow * LDS_STRIDE + akc * 8]);
    // Wt tile: 512 rows(N) x 32 K (8 b128 per thread)
    #pragma unroll
    for (int i = 0; i < 8; ++i) {
      int chunk = tid + i * 256;
      int n  = chunk >> 2;
      int kc = chunk & 3;
      stage16(Wt + (size_t)n * DD + k0 + kc * 8,
              &Ws[b][n * LDS_STRIDE + kc * 8]);
    }
  };

  stage(0, 0);

  int buf = 0;
  for (int k0 = 0; k0 < DD; k0 += 32, buf ^= 1) {
#if HAVE_ASYNC_LDS
    __builtin_amdgcn_s_wait_asynccnt(0);
#endif
    __syncthreads();                     // staged tiles visible to all waves
    if (k0 + 32 < DD) stage(buf ^ 1, k0 + 32);   // overlap next fill

    v16bf af[4], bfr[4];
    #pragma unroll
    for (int mt = 0; mt < 4; ++mt)
      af[mt] = load_frag_lds(&As[buf][0], mt * 16 + col16, half);
    #pragma unroll
    for (int nt = 0; nt < 4; ++nt)
      bfr[nt] = load_frag_lds(&Ws[buf][0], wave * 64 + nt * 16 + col16, half);

    #pragma unroll
    for (int mt = 0; mt < 4; ++mt)
      #pragma unroll
      for (int nt = 0; nt < 4; ++nt)
        acc[mt][nt] = __builtin_amdgcn_wmma_f32_16x16x32_bf16(
            false, af[mt], false, bfr[nt], (short)0, acc[mt][nt],
            false, false);
    // Per-wave s_wait_dscnt before each WMMA guarantees this wave's LDS reads
    // of `buf` completed before it reaches the next barrier, so `buf` may be
    // overwritten after that barrier.
  }
  __syncthreads();

  // ---- Epilogue: bias + GELU ----
  float bv[4];
  #pragma unroll
  for (int nt = 0; nt < 4; ++nt)
    bv[nt] = bias[wave * 64 + nt * 16 + col16];

  #pragma unroll
  for (int mt = 0; mt < 4; ++mt)
    #pragma unroll
    for (int nt = 0; nt < 4; ++nt)
      #pragma unroll
      for (int g = 0; g < 8; ++g)
        acc[mt][nt][g] = gelu_tanh(acc[mt][nt][g] + bv[nt]);

  // ---- LN statistics: per-row sums across this wave's 64 N, reduce across
  //      the 16 lanes of each half (halves hold different M rows), then
  //      accumulate into LDS with ds_add_f32 atomics. ----
  #pragma unroll
  for (int mt = 0; mt < 4; ++mt) {
    #pragma unroll
    for (int g = 0; g < 8; ++g) {
      float s = 0.f, ss = 0.f;
      #pragma unroll
      for (int nt = 0; nt < 4; ++nt) {
        float v = acc[mt][nt][g];
        s += v; ss += v * v;
      }
      #pragma unroll
      for (int off = 8; off >= 1; off >>= 1) {
        s  += __shfl_xor(s,  off, 32);
        ss += __shfl_xor(ss, off, 32);
      }
      if (col16 == 0) {
        int m = mt * 16 + g + half * 8;
        atomicAdd(&rsum[m], s);
        atomicAdd(&rssq[m], ss);
      }
    }
  }
  __syncthreads();

  // ---- Normalize + store bf16 ----
  #pragma unroll
  for (int mt = 0; mt < 4; ++mt) {
    #pragma unroll
    for (int g = 0; g < 8; ++g) {
      int m = mt * 16 + g + half * 8;
      float mean = rsum[m] * (1.f / (float)HH);
      float var  = rssq[m] * (1.f / (float)HH) - mean * mean;
      float rstd = rsqrtf(var + LN_EPS);
      size_t grow = (size_t)(m0 + m) * HH;
      #pragma unroll
      for (int nt = 0; nt < 4; ++nt) {
        float v = (acc[mt][nt][g] - mean) * rstd;
        out[grow + wave * 64 + nt * 16 + col16] = __float2bfloat16(v);
      }
    }
  }
}

// ---------------------------------------------------------------------------
// Kernel 4: logits[row] = h2[row,:] . Wa + ba   (one wave per row)
// ---------------------------------------------------------------------------
__global__ void logits_kernel(const __hip_bfloat16* __restrict__ h,
                              const float* __restrict__ Wa,
                              const float* __restrict__ ba,
                              float* __restrict__ logits) {
  int wave = threadIdx.x >> 5, lane = threadIdx.x & 31;
  int row = blockIdx.x * 8 + wave;
  const __hip_bfloat16* hr = h + (size_t)row * HH;
  float acc = 0.f;
  #pragma unroll
  for (int i = 0; i < 16; ++i) {
    int k = i * 32 + lane;
    acc = fmaf(__bfloat162float(hr[k]), Wa[k], acc);
  }
  #pragma unroll
  for (int off = 16; off >= 1; off >>= 1) acc += __shfl_down(acc, off, 32);
  if (lane == 0) logits[row] = acc + ba[0];
}

// ---------------------------------------------------------------------------
// Kernel 5: softmax over S=2048 per batch row (one block per b)
// ---------------------------------------------------------------------------
__global__ void softmax_kernel(const float* __restrict__ logits,
                               float* __restrict__ alpha) {
  int b = blockIdx.x, tid = threadIdx.x;
  __shared__ float red[256];
  const float* lr = logits + (size_t)b * SS;
  float v[8];
  float mx = -3.4e38f;
  #pragma unroll
  for (int e = 0; e < 8; ++e) { v[e] = lr[tid + e * 256]; mx = fmaxf(mx, v[e]); }
  red[tid] = mx; __syncthreads();
  for (int off = 128; off >= 1; off >>= 1) {
    if (tid < off) red[tid] = fmaxf(red[tid], red[tid + off]);
    __syncthreads();
  }
  mx = red[0]; __syncthreads();
  float s = 0.f;
  #pragma unroll
  for (int e = 0; e < 8; ++e) { v[e] = __expf(v[e] - mx); s += v[e]; }
  red[tid] = s; __syncthreads();
  for (int off = 128; off >= 1; off >>= 1) {
    if (tid < off) red[tid] += red[tid + off];
    __syncthreads();
  }
  float inv = 1.f / red[0];
  #pragma unroll
  for (int e = 0; e < 8; ++e) alpha[(size_t)b * SS + tid + e * 256] = v[e] * inv;
}

// ---------------------------------------------------------------------------
// Kernel 6: eps = LN(sigmoid(alpha @ We + be)); mask = alpha > eps
//           one block per batch row (block owns the whole 2048-wide row)
// ---------------------------------------------------------------------------
__global__ void eps_mask_kernel(const float* __restrict__ alpha,
                                const float* __restrict__ We,
                                const float* __restrict__ be,
                                int* __restrict__ mask) {
  int b = blockIdx.x, tid = threadIdx.x;
  __shared__ float sa[SS];
  __shared__ float red[256];
  #pragma unroll
  for (int e = 0; e < 8; ++e) sa[tid + e * 256] = alpha[(size_t)b * SS + tid + e * 256];
  __syncthreads();

  float acc[8];
  #pragma unroll
  for (int e = 0; e < 8; ++e) acc[e] = 0.f;
  for (int s = 0; s < SS; ++s) {
    float av = sa[s];
    const float* wr = We + (size_t)s * SS + tid;
    #pragma unroll
    for (int e = 0; e < 8; ++e) acc[e] = fmaf(av, wr[e * 256], acc[e]);
  }

  float sg[8], ls = 0.f, lss = 0.f;
  #pragma unroll
  for (int e = 0; e < 8; ++e) {
    float t = acc[e] + be[tid + e * 256];
    float x = 1.f / (1.f + __expf(-t));
    sg[e] = x; ls += x; lss += x * x;
  }
  red[tid] = ls; __syncthreads();
  for (int off = 128; off >= 1; off >>= 1) {
    if (tid < off) red[tid] += red[tid + off];
    __syncthreads();
  }
  ls = red[0]; __syncthreads();
  red[tid] = lss; __syncthreads();
  for (int off = 128; off >= 1; off >>= 1) {
    if (tid < off) red[tid] += red[tid + off];
    __syncthreads();
  }
  lss = red[0];
  float mean = ls * (1.f / (float)SS);
  float var  = lss * (1.f / (float)SS) - mean * mean;
  float rstd = rsqrtf(var + LN_EPS);
  #pragma unroll
  for (int e = 0; e < 8; ++e) {
    int c = tid + e * 256;
    float ev = (sg[e] - mean) * rstd;
    mask[(size_t)b * SS + c] = (sa[c] > ev) ? 1 : 0;
  }
}

// ---------------------------------------------------------------------------
// Kernel 7: stable compaction -> order (int32) + valid (one block per b)
// ---------------------------------------------------------------------------
__global__ void order_kernel(const int* __restrict__ mask,
                             int* __restrict__ order,
                             float* __restrict__ valid) {
  int b = blockIdx.x, tid = threadIdx.x;
  __shared__ int sc[256];
  int m[8], cnt = 0;
  const int* mr = mask + (size_t)b * SS + tid * 8;
  #pragma unroll
  for (int e = 0; e < 8; ++e) { m[e] = mr[e]; cnt += m[e]; }
  sc[tid] = cnt; __syncthreads();
  for (int off = 1; off < 256; off <<= 1) {
    int v = (tid >= off) ? sc[tid - off] : 0;
    __syncthreads();
    sc[tid] += v;
    __syncthreads();
  }
  int total = sc[255];
  int run = sc[tid] - cnt;                 // #selected before my chunk
  int* orow = order + (size_t)b * SS;
  #pragma unroll
  for (int e = 0; e < 8; ++e) {
    int i = tid * 8 + e;
    if (m[e]) { orow[run] = i; run++; }
    else      { orow[total + i - run] = i; }
  }
  float* vrow = valid + (size_t)b * SS;
  #pragma unroll
  for (int e = 0; e < 8; ++e) {
    int j = tid * 8 + e;
    vrow[j] = (j < total) ? 1.f : 0.f;
  }
}

// ---------------------------------------------------------------------------
// Kernel 8: gathered[b][j][:] = x[b][order[b][j]][:]   (float4 copies)
// ---------------------------------------------------------------------------
__global__ void gather_kernel(const float* __restrict__ x,
                              const int* __restrict__ order,
                              float* __restrict__ gathered) {
  int blk = blockIdx.x;
  int b = blk >> 11, j = blk & 2047;
  int src = order[(size_t)b * SS + j];
  const float4* xs = (const float4*)(x + ((size_t)b * SS + src) * DD);
  float4* gd = (float4*)(gathered + (size_t)blk * DD);
  gd[threadIdx.x] = xs[threadIdx.x];
}

// ---------------------------------------------------------------------------
// Host launcher
// ---------------------------------------------------------------------------
extern "C" void kernel_launch(void* const* d_in, const int* in_sizes, int n_in,
                              void* d_out, int out_size, void* d_ws, size_t ws_size,
                              hipStream_t stream) {
  const float* x  = (const float*)d_in[0];
  const float* W0 = (const float*)d_in[1];
  const float* b0 = (const float*)d_in[2];
  const float* W1 = (const float*)d_in[3];
  const float* b1 = (const float*)d_in[4];
  const float* Wa = (const float*)d_in[5];
  const float* ba = (const float*)d_in[6];
  const float* We = (const float*)d_in[7];
  const float* be = (const float*)d_in[8];

  // Workspace layout (256B-aligned slabs)
  size_t off = 0;
  char* wsb = (char*)d_ws;
  auto take = [&](size_t bytes) -> void* {
    void* p = wsb + off;
    off += (bytes + 255) & ~(size_t)255;
    return p;
  };
  __hip_bfloat16* xb  = (__hip_bfloat16*)take((size_t)MROWS * DD * 2);
  __hip_bfloat16* Wt0 = (__hip_bfloat16*)take((size_t)DD * HH * 2);
  __hip_bfloat16* Wt1 = (__hip_bfloat16*)take((size_t)HH * HH * 2);
  __hip_bfloat16* h1  = (__hip_bfloat16*)take((size_t)MROWS * HH * 2);
  __hip_bfloat16* h2  = (__hip_bfloat16*)take((size_t)MROWS * HH * 2);
  float* logits = (float*)take((size_t)MROWS * 4);
  float* alpha  = (float*)take((size_t)BB * SS * 4);
  int*   mask   = (int*)take((size_t)BB * SS * 4);

  // Output regions (d_out treated as array of 4-byte elements)
  int*   out_order    = (int*)d_out;
  float* out_gathered = (float*)d_out + (size_t)BB * SS;
  float* out_valid    = (float*)d_out + (size_t)BB * SS + (size_t)BB * SS * DD;

  // 1. x -> bf16
  cvt_f32_bf16_kernel<<<(MROWS * DD) / 1024, 256, 0, stream>>>(x, xb);
  // 2. W0, W1 -> transposed bf16
  transpose_cvt_kernel<<<dim3(HH / 32, DD / 32), 256, 0, stream>>>(W0, Wt0);
  transpose_cvt_kernel<<<dim3(HH / 32, HH / 32), 256, 0, stream>>>(W1, Wt1);
  // 3. h1 = LN(gelu(x @ W0 + b0)),  h2 = LN(gelu(h1 @ W1 + b1))
  gemm_gelu_ln_kernel<<<MROWS / 64, 256, 0, stream>>>(xb, Wt0, b0, h1);
  gemm_gelu_ln_kernel<<<MROWS / 64, 256, 0, stream>>>(h1, Wt1, b1, h2);
  // 4. logits = h2 . Wa + ba
  logits_kernel<<<MROWS / 8, 256, 0, stream>>>(h2, Wa, ba, logits);
  // 5. alpha = softmax(logits, axis=S)
  softmax_kernel<<<BB, 256, 0, stream>>>(logits, alpha);
  // 6. eps = LN(sigmoid(alpha @ We + be)); mask = alpha > eps
  eps_mask_kernel<<<BB, 256, 0, stream>>>(alpha, We, be, mask);
  // 7. order + valid
  order_kernel<<<BB, 256, 0, stream>>>(mask, out_order, out_valid);
  // 8. gathered
  gather_kernel<<<BB * SS, 128, 0, stream>>>(x, out_order, out_gathered);
}
